// CTRNN_25666724561014
// MI455X (gfx1250) — matrix operand order repaired and checked by reading, not compiled, not run on hardware
//
#include <hip/hip_runtime.h>

// CTRNN on gfx1250 (MI455X).
//
// Phase 0: split W_hh (fp32) into bf16 hi/lo pair in d_ws (bf16x3 scheme).
// Phase 1: xp = x @ W_in^T + b_in + b_hh  -> f32 WMMA (16x16x4), memory bound,
//          full-chip parallel; xp stored fp32 in d_ws (512 MB).
// Phase 2: ONE persistent scan kernel. The recurrence is batch-row
//          independent, so each block owns 16 batch rows for all 1024 steps:
//          h state in registers (f32, D-fragment layout), bf16 hi/lo copy of
//          h double-buffered in LDS for the K-dim exchange across waves,
//          one workgroup barrier per step. GEMM = bf16x3 split-precision
//          (a_hi*w_hi + a_lo*w_hi + a_hi*w_lo), f32 accumulate -> fp32-class
//          accuracy at bf16-WMMA throughput. 4 replicas per M-tile spread the
//          1 GB of output/speed stores over 64 WGPs.

#define SEQ   1024
#define BATCH 256
#define IN_F  256
#define HID   512
#define ALPHA 0.2f
#define R_REP 4

typedef __attribute__((ext_vector_type(2)))  float   v2f;
typedef __attribute__((ext_vector_type(8)))  float   v8f;
typedef __attribute__((ext_vector_type(8)))  __bf16  v8bf;
typedef __attribute__((ext_vector_type(16))) __bf16  v16bf;

// ---- Phase 0: split W_hh into bf16 hi + residual lo ---------------------
__global__ __launch_bounds__(256) void split_whh(const float* __restrict__ w,
                                                 __bf16* __restrict__ hi,
                                                 __bf16* __restrict__ lo) {
  const int i = blockIdx.x * 256 + threadIdx.x;
  const float f = w[i];
  const __bf16 h = (__bf16)f;          // RNE
  hi[i] = h;
  lo[i] = (__bf16)(f - (float)h);      // residual
}

// ---- f32 WMMA fragment loader (A 16x4 / B 4x16 layouts coincide) --------
__device__ __forceinline__ v2f ld_frag_f32(const float* __restrict__ src,
                                           int ld, int rc0, int kbase,
                                           int lane) {
  const int rc = rc0 + (lane & 15);
  const int k  = kbase + ((lane >> 4) << 1);
  const float* p = src + (size_t)rc * ld + k;
  v2f f;
  f.x = p[0];
  f.y = p[1];
  return f;
}

// ---- bf16 A fragment from LDS (16-bit A 16x32 layout) -------------------
// lane r holds row r; elements 0..15 <-> K = kb + 16*half + 0..15 (contiguous)
__device__ __forceinline__ v16bf ld_a_lds(const __bf16* row, int kb, int half) {
  return *(const v16bf*)(row + kb + 16 * half);
}

// ---- Phase 1: input projection GEMM (f32 WMMA) --------------------------
__global__ __launch_bounds__(256) void ctrnn_xp_gemm(
    const float* __restrict__ x,     // (SEQ*BATCH, IN)
    const float* __restrict__ W_in,  // (HID, IN)
    const float* __restrict__ b_in,
    const float* __restrict__ b_hh,
    float* __restrict__ xp)          // (SEQ*BATCH, HID)
{
  const int lane = threadIdx.x & 31;
  const int wave = threadIdx.x >> 5;   // 0..7
  const int m0   = blockIdx.x * 16;
  const int n0w  = wave * 64;

  v8f acc0 = {}, acc1 = {}, acc2 = {}, acc3 = {};
  for (int kb = 0; kb < IN_F; kb += 4) {
    v2f a  = ld_frag_f32(x, IN_F, m0, kb, lane);
    v2f b0 = ld_frag_f32(W_in, IN_F, n0w + 0, kb, lane);
    v2f b1 = ld_frag_f32(W_in, IN_F, n0w + 16, kb, lane);
    v2f b2 = ld_frag_f32(W_in, IN_F, n0w + 32, kb, lane);
    v2f b3 = ld_frag_f32(W_in, IN_F, n0w + 48, kb, lane);
    acc0 = __builtin_amdgcn_wmma_f32_16x16x4_f32(false, a, false, b0,
                                                 (short)0, acc0, false, false);
    acc1 = __builtin_amdgcn_wmma_f32_16x16x4_f32(false, a, false, b1,
                                                 (short)0, acc1, false, false);
    acc2 = __builtin_amdgcn_wmma_f32_16x16x4_f32(false, a, false, b2,
                                                 (short)0, acc2, false, false);
    acc3 = __builtin_amdgcn_wmma_f32_16x16x4_f32(false, a, false, b3,
                                                 (short)0, acc3, false, false);
  }

  const int half = lane >> 4;
  v8f accs[4] = {acc0, acc1, acc2, acc3};
#pragma unroll
  for (int s = 0; s < 4; ++s) {
    const int n = n0w + s * 16 + (lane & 15);
    const float bias = b_in[n] + b_hh[n];
#pragma unroll
    for (int j = 0; j < 8; ++j) {
      const int m = m0 + j + 8 * half;
      xp[(size_t)m * HID + n] = accs[s][j] + bias;
    }
  }
}

// ---- Phase 2: persistent scan (bf16x3 WMMA) -----------------------------
// grid = 16 M-tiles * R_REP replicas; block = 256 (8 waves).
// Wave w computes columns [64w, 64w+64); replica r's waves 2r,2r+1 store
// columns [128r, 128r+128) to global.
__global__ __launch_bounds__(256) void ctrnn_scan(
    const float*  __restrict__ xp,    // (SEQ*BATCH, HID)
    const __bf16* __restrict__ w_hi,  // (HID, HID)
    const __bf16* __restrict__ w_lo,  // (HID, HID)
    float* __restrict__ out_base)     // d_out: [output | h_last | speed]
{
  __shared__ __bf16 h_hi[2][16][HID];   // 32 KB, double-buffered
  __shared__ __bf16 h_lo[2][16][HID];   // 32 KB

  const int lane = threadIdx.x & 31;
  const int wave = threadIdx.x >> 5;
  const int mtile = blockIdx.x / R_REP;
  const int rep   = blockIdx.x % R_REP;
  const int m0   = mtile * 16;
  const int n0w  = wave * 64;
  const int r_a  = lane & 15;
  const int half = lane >> 4;

  const size_t BH = (size_t)BATCH * HID;
  float* output = out_base;
  float* h_last = out_base + (size_t)SEQ * BH;
  float* speed  = h_last + BH;

  // h(0) == 0: zero both LDS buffers' slot 0 and the register state.
  for (int i = threadIdx.x; i < 16 * HID; i += 256) {
    (&h_hi[0][0][0])[i] = (__bf16)0.0f;
    (&h_lo[0][0][0])[i] = (__bf16)0.0f;
  }
  v8f hst[4] = {};
  __syncthreads();

  const bool store_g = ((wave >> 1) == rep);

#pragma unroll 1
  for (int t = 0; t < SEQ; ++t) {
    const int cur = t & 1;
    const int nxt = cur ^ 1;

    v8f acc[4] = {};
#pragma unroll 4
    for (int kb = 0; kb < HID; kb += 32) {
      const v16bf ahi = ld_a_lds(&h_hi[cur][r_a][0], kb, half);
      const v16bf alo = ld_a_lds(&h_lo[cur][r_a][0], kb, half);
      v16bf bhi[4], blo[4];
#pragma unroll
      for (int s = 0; s < 4; ++s) {
        // B 32x16 layout: lane n holds K = kb + 16*half + 0..15 (contiguous)
        const size_t woff =
            (size_t)(n0w + s * 16 + r_a) * HID + kb + 16 * half;
        bhi[s] = *(const v16bf*)(w_hi + woff);
        blo[s] = *(const v16bf*)(w_lo + woff);
      }
#pragma unroll
      for (int s = 0; s < 4; ++s)
        acc[s] = __builtin_amdgcn_wmma_f32_16x16x32_bf16(
            false, ahi, false, bhi[s], (short)0, acc[s], false, false);
#pragma unroll
      for (int s = 0; s < 4; ++s)
        acc[s] = __builtin_amdgcn_wmma_f32_16x16x32_bf16(
            false, alo, false, bhi[s], (short)0, acc[s], false, false);
#pragma unroll
      for (int s = 0; s < 4; ++s)
        acc[s] = __builtin_amdgcn_wmma_f32_16x16x32_bf16(
            false, ahi, false, blo[s], (short)0, acc[s], false, false);
    }

    const float* xp_t = xp + (size_t)t * BH;
    float* out_t = output + (size_t)t * BH;
    float* spd_t = speed + (size_t)t * BH;
    const bool last = (t == SEQ - 1);

#pragma unroll
    for (int s = 0; s < 4; ++s) {
      const int n = n0w + s * 16 + r_a;
#pragma unroll
      for (int j = 0; j < 8; ++j) {
        const int ml = j + 8 * half;
        const size_t idx = (size_t)(m0 + ml) * HID + n;
        const float h = hst[s][j];
        float pre = xp_t[idx] + acc[s][j];
        pre = pre > 0.0f ? pre : 0.0f;
        const float hn = h * (1.0f - ALPHA) + pre * ALPHA;
        hst[s][j] = hn;
        const __bf16 hb = (__bf16)hn;
        h_hi[nxt][ml][n] = hb;
        h_lo[nxt][ml][n] = (__bf16)(hn - (float)hb);
        if (store_g) {
          out_t[idx] = hn;
          spd_t[idx] = hn - h;
          if (last) h_last[idx] = hn;
        }
      }
    }
    __syncthreads();  // h[nxt] fully written before next step reads it
  }
}

extern "C" void kernel_launch(void* const* d_in, const int* in_sizes, int n_in,
                              void* d_out, int out_size, void* d_ws,
                              size_t ws_size, hipStream_t stream) {
  (void)in_sizes; (void)n_in; (void)out_size; (void)ws_size;
  const float* x    = (const float*)d_in[0];  // (SEQ, BATCH, IN)
  const float* W_in = (const float*)d_in[1];  // (HID, IN)
  const float* b_in = (const float*)d_in[2];  // (HID)
  const float* W_hh = (const float*)d_in[3];  // (HID, HID)
  const float* b_hh = (const float*)d_in[4];  // (HID)
  float* out = (float*)d_out;

  // Workspace layout: [w_hi 512KB | w_lo 512KB | xp 512MB]
  __bf16* w_hi = (__bf16*)d_ws;
  __bf16* w_lo = w_hi + (size_t)HID * HID;
  float*  xp   = (float*)((char*)d_ws + 2ull * HID * HID * sizeof(__bf16));

  split_whh<<<(HID * HID) / 256, 256, 0, stream>>>(W_hh, w_hi, w_lo);
  ctrnn_xp_gemm<<<(SEQ * BATCH) / 16, 256, 0, stream>>>(x, W_in, b_in, b_hh, xp);
  ctrnn_scan<<<16 * R_REP, 256, 0, stream>>>(xp, w_hi, w_lo, out);
}
